// DeformConv_90709709291660
// MI455X (gfx1250) — matrix operand order repaired and checked by reading, not compiled, not run on hardware
//
#include <hip/hip_runtime.h>
#include <math.h>

typedef __bf16 bf16;
typedef __attribute__((ext_vector_type(16))) __bf16 v16bf;
typedef __attribute__((ext_vector_type(8)))  __bf16 v8bf;
typedef __attribute__((ext_vector_type(8)))  float  v8f;

#define HH_  96
#define WW_  96
#define CIN  256
#define COUT 256
#define HW_  (HH_*WW_)
#define NITER 72          // 2304 / 32
#define WT_ELEMS   (9*256*256)
#define WOFF_ELEMS (9*32*256)
#define WOFF_OFF_BYTES (WT_ELEMS*2)

// ---------- weight reorder: w(O,C,3,3) f32 -> wT[k][n][c] bf16 -----------
__global__ void reorder_w_kernel(const float* __restrict__ w, bf16* __restrict__ wT) {
  int t = blockIdx.x * blockDim.x + threadIdx.x;
  if (t >= WT_ELEMS) return;
  int k = t / 65536, rem = t % 65536, n = rem / 256, c = rem % 256;
  wT[t] = (bf16)w[n * 2304 + c * 9 + k];
}

// ---- offset-weight reorder: w_off(27,C,3,3) f32 -> [k][n(32,pad)][c] bf16 ----
__global__ void reorder_woff_kernel(const float* __restrict__ w, bf16* __restrict__ wT) {
  int t = blockIdx.x * blockDim.x + threadIdx.x;
  if (t >= WOFF_ELEMS) return;
  int k = t / 8192, rem = t % 8192, n = rem / 256, c = rem % 256;
  wT[t] = (n < 27) ? (bf16)w[n * 2304 + c * 9 + k] : (bf16)0.0f;
}

__device__ __forceinline__ int clampi(int v, int lo, int hi) {
  return v < lo ? lo : (v > hi ? hi : v);
}

__device__ __forceinline__ v16bf mkfrag(v8bf lo, v8bf hi) {
  return __builtin_shufflevector(lo, hi, 0,1,2,3,4,5,6,7,8,9,10,11,12,13,14,15);
}

// One workgroup = 16 spatial positions x ALL 256 output channels (16 waves).
__launch_bounds__(512)
__global__ void dcn_fused_kernel(const float* __restrict__ x,
                                 const bf16*  __restrict__ wT,
                                 const bf16*  __restrict__ wOffT,
                                 const float* __restrict__ b_off,
                                 const float* __restrict__ bconv,
                                 const float* __restrict__ gamma,
                                 const float* __restrict__ beta,
                                 const float* __restrict__ rmean,
                                 const float* __restrict__ rvar,
                                 float* __restrict__ out)
{
  __shared__ __align__(16) bf16 As1[8][16][32];   // phase-1 im2col: 8 c-chunks
  __shared__ float offAcc[8][16][32];             // phase-1 partial accumulators
  __shared__ __align__(16) bf16 As2[2][16][32];   // phase-2 A tile, double-buffered
  __shared__ float dyL[9][16], dxL[9][16], mkL[9][16];

  const int tid    = (int)threadIdx.x;
  const int wave   = tid >> 5;
  const int lane   = tid & 31;
  const int half   = lane >> 4;
  const int lane16 = lane & 15;

  const int m0  = (int)blockIdx.x * 16;
  const int b   = m0 / HW_;
  const int hw0 = m0 % HW_;
  const int hh  = hw0 / WW_;          // same row for all 16 m (96 % 16 == 0)
  const int ww0 = hw0 % WW_;
  const float* __restrict__ xb = x + (size_t)b * CIN * HW_;

  // A-tile builder mapping: thread -> (m, channel-in-chunk)
  const int am = tid & 15;
  const int aj = tid >> 4;            // 0..31
  const int aw = ww0 + am;

  // =============== Phase 1: offset conv, K split across waves ==============
  // wave -> (c-chunk pair p1 = wave/2, n-half sub1 = wave&1); 9 tap-steps.
  v8f acc1 = {};
  const int p1   = wave >> 1;
  const int sub1 = wave & 1;
  for (int s = 0; s < 9; ++s) {
    const int ky = s / 3 - 1, kx = s % 3 - 1;
    const int sy = hh + ky, sx = aw + kx;
    // branchless: clamp address, zero via multiply
    const float inb = (sy >= 0 && sy < HH_ && sx >= 0 && sx < WW_) ? 1.0f : 0.0f;
    const float* __restrict__ xp =
        xb + clampi(sy, 0, HH_ - 1) * WW_ + clampi(sx, 0, WW_ - 1);
#pragma unroll
    for (int p = 0; p < 8; ++p) {
      const int c = p * 32 + aj;
      As1[p][am][aj] = (bf16)(xp[(size_t)c * HW_] * inb);
    }
    __syncthreads();
    {
      v8bf alo = *(const v8bf*)&As1[p1][lane16][half * 8];
      v8bf ahi = *(const v8bf*)&As1[p1][lane16][16 + half * 8];
      v16bf afr = mkfrag(alo, ahi);
      const bf16* bp = wOffT + (((size_t)s * 32 + sub1 * 16 + lane16) * 256
                                + p1 * 32 + half * 8);
      v16bf bfr = mkfrag(*(const v8bf*)bp, *(const v8bf*)(bp + 16));
      acc1 = __builtin_amdgcn_wmma_f32_16x16x32_bf16(false, afr, false, bfr,
                                                     (short)0, acc1, false, false);
    }
    __syncthreads();
  }
  // dump per-wave partials, then reduce 8 c-chunk partials per (m,n)
  {
    const int nn = sub1 * 16 + lane16;
#pragma unroll
    for (int g = 0; g < 8; ++g) offAcc[p1][g + half * 8][nn] = acc1[g];
  }
  __syncthreads();
  {
    const int mm = tid >> 5, nn = tid & 31;
    float v = 0.0f;
#pragma unroll
    for (int p = 0; p < 8; ++p) v += offAcc[p][mm][nn];
    if (nn < 18) {
      v += b_off[nn];
      if ((nn & 1) == 0) dyL[nn >> 1][mm] = v;
      else               dxL[nn >> 1][mm] = v;
    } else if (nn < 27) {
      v += b_off[nn];
      mkL[nn - 18][mm] = 1.0f / (1.0f + __expf(-v));
    }
  }
  __syncthreads();

  // ========= Phase 2: modulated deformable GEMM, double-buffered ==========
  v8f acc = {};
  const int n0 = wave << 4;

  auto buildA = [&](int it, int bufw) {
    const int k  = it >> 3;
    const int c0 = (it & 7) << 5;
    const int ky = k / 3 - 1, kx = k % 3 - 1;
    const float dy = dyL[k][am], dx = dxL[k][am], mk = mkL[k][am];
    const float py = (float)(hh + ky) + dy;
    const float px = (float)(aw + kx) + dx;
    const float y0f = floorf(py), x0f = floorf(px);
    const float ly = py - y0f, lx = px - x0f;
    const int y0 = (int)y0f, x0 = (int)x0f;
    // branchless gathers: clamped addresses, validity folded into weights
    const int y0c = clampi(y0,     0, HH_ - 1);
    const int y1c = clampi(y0 + 1, 0, HH_ - 1);
    const int x0c = clampi(x0,     0, WW_ - 1);
    const int x1c = clampi(x0 + 1, 0, WW_ - 1);
    const float vy0 = (y0     >= 0 && y0     < HH_) ? 1.0f : 0.0f;
    const float vy1 = (y0 + 1 >= 0 && y0 + 1 < HH_) ? 1.0f : 0.0f;
    const float vx0 = (x0     >= 0 && x0     < WW_) ? 1.0f : 0.0f;
    const float vx1 = (x0 + 1 >= 0 && x0 + 1 < WW_) ? 1.0f : 0.0f;
    const float* __restrict__ xc = xb + (size_t)(c0 + aj) * HW_;
    const float v00 = xc[y0c * WW_ + x0c];
    const float v01 = xc[y0c * WW_ + x1c];
    const float v10 = xc[y1c * WW_ + x0c];
    const float v11 = xc[y1c * WW_ + x1c];
    const float w00 = (1.0f - ly) * (1.0f - lx) * vy0 * vx0;
    const float w01 = (1.0f - ly) * lx          * vy0 * vx1;
    const float w10 = ly          * (1.0f - lx) * vy1 * vx0;
    const float w11 = ly          * lx          * vy1 * vx1;
    const float v = (v00 * w00 + v01 * w01 + v10 * w10 + v11 * w11) * mk;
    As2[bufw][am][aj] = (bf16)v;
  };

  buildA(0, 0);
  __syncthreads();
  int buf = 0;
  for (int it = 0; it < NITER; ++it) {
    if (it + 1 < NITER) buildA(it + 1, buf ^ 1);   // overlap gathers with WMMA
    {
      const int k  = it >> 3;
      const int c0 = (it & 7) << 5;
      v8bf alo = *(const v8bf*)&As2[buf][lane16][half * 8];
      v8bf ahi = *(const v8bf*)&As2[buf][lane16][16 + half * 8];
      v16bf afr = mkfrag(alo, ahi);
      const bf16* bp = wT + (((size_t)k * 256 + n0 + lane16) * 256 + c0 + half * 8);
      v16bf bfr = mkfrag(*(const v8bf*)bp, *(const v8bf*)(bp + 16));
      acc = __builtin_amdgcn_wmma_f32_16x16x32_bf16(false, afr, false, bfr,
                                                    (short)0, acc, false, false);
    }
    __syncthreads();
    buf ^= 1;
  }

  // ===================== epilogue: BN + ReLU + store =====================
  {
    const int nn = n0 + lane16;
    const float sc = gamma[nn] * rsqrtf(rvar[nn] + 1e-5f);
    const float bs = beta[nn] - rmean[nn] * sc;
    const float bc = bconv[nn];
    float* __restrict__ op = out + ((size_t)b * COUT + nn) * HW_ + hw0;
#pragma unroll
    for (int g = 0; g < 8; ++g) {
      const int mm = g + half * 8;
      float v = (acc[g] + bc) * sc + bs;
      op[mm] = v > 0.0f ? v : 0.0f;
    }
  }
}

extern "C" void kernel_launch(void* const* d_in, const int* in_sizes, int n_in,
                              void* d_out, int out_size, void* d_ws, size_t ws_size,
                              hipStream_t stream) {
  (void)in_sizes; (void)n_in; (void)out_size; (void)ws_size;
  const float* x     = (const float*)d_in[0];
  const float* w_off = (const float*)d_in[1];
  const float* b_off = (const float*)d_in[2];
  const float* w     = (const float*)d_in[3];
  const float* bconv = (const float*)d_in[4];
  const float* gamma = (const float*)d_in[5];
  const float* beta  = (const float*)d_in[6];
  const float* rmean = (const float*)d_in[7];
  const float* rvar  = (const float*)d_in[8];
  float* out = (float*)d_out;

  bf16* wT    = (bf16*)d_ws;
  bf16* wOffT = (bf16*)((char*)d_ws + WOFF_OFF_BYTES);

  reorder_w_kernel<<<(WT_ELEMS + 255) / 256, 256, 0, stream>>>(w, wT);
  reorder_woff_kernel<<<(WOFF_ELEMS + 255) / 256, 256, 0, stream>>>(w_off, wOffT);

  // 36864 positions / 16 per block = 2304 blocks, 512 threads (16 waves) each
  dcn_fused_kernel<<<2304, 512, 0, stream>>>(x, wT, wOffT, b_off, bconv,
                                             gamma, beta, rmean, rvar, out);
}
